// MultiHeadAttention_71768903516708
// MI455X (gfx1250) — matrix operand order repaired and checked by reading, not compiled, not run on hardware
//
#include <hip/hip_runtime.h>

#define B_  4
#define S_  2048
#define H_  1024
#define NH_ 16
#define DH_ 64
#define FILLV (-1.0e13f)

typedef __attribute__((ext_vector_type(16))) __bf16 v16bf;
typedef __attribute__((ext_vector_type(2)))  __bf16 v2bf;
typedef __attribute__((ext_vector_type(8)))  float  v8f;
typedef int v4i_vs __attribute__((vector_size(16)));   // matches builtin's int4

union BF16x16 { uint4 u[2]; v16bf v; };

// ---- f32 -> bf16 conversion ----
// Preferred: single V_CVT_PK_BF16_F32 (not present on this toolchain's gfx1250).
// Fallback: +0x8000 round-to-nearest bias, then one V_PERM_B32 splices the two
// high halves: [b.hi16 | a.hi16]. 3 VALU per pair vs ~7 for the manual RNE.
#if __has_builtin(__builtin_amdgcn_cvt_pk_bf16_f32)
__device__ __forceinline__ unsigned pack2bf(float a, float b) {
  union { v2bf v; unsigned u; } c;
  c.v = __builtin_amdgcn_cvt_pk_bf16_f32(a, b);
  return c.u;
}
#else
__device__ __forceinline__ unsigned pack2bf(float a, float b) {
  const unsigned ua = __float_as_uint(a) + 0x8000u;
  const unsigned ub = __float_as_uint(b) + 0x8000u;
  // v_perm_b32: sel bytes 0-3 pick from arg1(S1), 4-7 from arg0(S0)
  return __builtin_amdgcn_perm(ub, ua, 0x07060302u);
}
#endif
__device__ __forceinline__ unsigned short f2bf(float f) {
  return (unsigned short)((__float_as_uint(f) + 0x8000u) >> 16);
}

// ---- CDNA5 async global->LDS copy (ASYNCcnt), signature per round-2 probe ----
#if __has_builtin(__builtin_amdgcn_global_load_async_to_lds_b128) && \
    __has_builtin(__builtin_amdgcn_s_wait_asynccnt)
#define USE_ASYNC_LDS 1
__device__ __forceinline__ void async_ld16(const void* g, void* l) {
  __builtin_amdgcn_global_load_async_to_lds_b128(
      (__attribute__((address_space(1))) v4i_vs*)g,
      (__attribute__((address_space(3))) v4i_vs*)l, 0, 0);
}
#else
#define USE_ASYNC_LDS 0
#endif

__device__ __forceinline__ v8f wmma_bf16(const BF16x16& a, const BF16x16& b, v8f c) {
  return __builtin_amdgcn_wmma_f32_16x16x32_bf16(false, a.v, false, b.v, (short)0, c, false, false);
}
__device__ __forceinline__ float rmax16(float x) {
  x = fmaxf(x, __shfl_xor(x, 1, 32));
  x = fmaxf(x, __shfl_xor(x, 2, 32));
  x = fmaxf(x, __shfl_xor(x, 4, 32));
  x = fmaxf(x, __shfl_xor(x, 8, 32));
  return x;
}
__device__ __forceinline__ float rsum16(float x) {
  x += __shfl_xor(x, 1, 32);
  x += __shfl_xor(x, 2, 32);
  x += __shfl_xor(x, 4, 32);
  x += __shfl_xor(x, 8, 32);
  return x;
}

// ---------------------------------------------------------------------------
// QKV projection: C[m,n] = X[m,:] . W[n,:] + bias[n]  (y = x @ W.T + b)
// Block tile 128x128, 256 thr = 8 waves (4 M x 2 N), wave tile 32x64.
// Register-prefetch of next global tile overlaps WMMA compute.
// Output bf16 scattered to [B,NH,S,DH] (vtrans=0) or [B,NH,DH,S] (vtrans=1).
// ---------------------------------------------------------------------------
__global__ __launch_bounds__(256) void qkv_proj_kernel(
    const float* __restrict__ X, const float* __restrict__ W,
    const float* __restrict__ bias, unsigned short* __restrict__ out, int vtrans)
{
  __shared__ __align__(16) unsigned short Xs[128][40];
  __shared__ __align__(16) unsigned short Ws[128][40];
  const int t = threadIdx.x;
  const int lane = t & 31, wave = t >> 5;
  const int wm = wave >> 1, wn = wave & 1;
  const int ln = lane & 15, lh = lane >> 4;
  const int bm = blockIdx.y * 128, bn = blockIdx.x * 128;

  const int row = t >> 1, seg = (t & 1) * 16;          // 16 floats per thread per tile
  const float* xb = X + (size_t)(bm + row) * H_ + seg;
  const float* wb = W + (size_t)(bn + row) * H_ + seg;

  const v8f zero = {0.f,0.f,0.f,0.f,0.f,0.f,0.f,0.f};
  v8f acc[2][4];
  #pragma unroll
  for (int fm = 0; fm < 2; ++fm)
    #pragma unroll
    for (int fn = 0; fn < 4; ++fn) acc[fm][fn] = zero;

  float4 x0, x1, x2, x3, w0, w1, w2, w3;
  {
    const float4* xp = (const float4*)xb;
    x0 = xp[0]; x1 = xp[1]; x2 = xp[2]; x3 = xp[3];
    const float4* wp = (const float4*)wb;
    w0 = wp[0]; w1 = wp[1]; w2 = wp[2]; w3 = wp[3];
  }

  for (int kt = 0; kt < H_; kt += 32) {
    __syncthreads();
    {
      unsigned* dx = (unsigned*)&Xs[row][seg];
      dx[0] = pack2bf(x0.x, x0.y); dx[1] = pack2bf(x0.z, x0.w);
      dx[2] = pack2bf(x1.x, x1.y); dx[3] = pack2bf(x1.z, x1.w);
      dx[4] = pack2bf(x2.x, x2.y); dx[5] = pack2bf(x2.z, x2.w);
      dx[6] = pack2bf(x3.x, x3.y); dx[7] = pack2bf(x3.z, x3.w);
      unsigned* dw = (unsigned*)&Ws[row][seg];
      dw[0] = pack2bf(w0.x, w0.y); dw[1] = pack2bf(w0.z, w0.w);
      dw[2] = pack2bf(w1.x, w1.y); dw[3] = pack2bf(w1.z, w1.w);
      dw[4] = pack2bf(w2.x, w2.y); dw[5] = pack2bf(w2.z, w2.w);
      dw[6] = pack2bf(w3.x, w3.y); dw[7] = pack2bf(w3.z, w3.w);
    }
    __syncthreads();
    if (kt + 32 < H_) {                                 // prefetch next tile
      const float4* xp = (const float4*)(xb + kt + 32);
      x0 = xp[0]; x1 = xp[1]; x2 = xp[2]; x3 = xp[3];
      const float4* wp = (const float4*)(wb + kt + 32);
      w0 = wp[0]; w1 = wp[1]; w2 = wp[2]; w3 = wp[3];
    }
    BF16x16 a[2], bb[4];
    #pragma unroll
    for (int fm = 0; fm < 2; ++fm) {
      const int r = wm * 32 + fm * 16 + ln;
      a[fm].u[0] = *(const uint4*)&Xs[r][lh * 8];
      a[fm].u[1] = *(const uint4*)&Xs[r][16 + lh * 8];
    }
    #pragma unroll
    for (int fn = 0; fn < 4; ++fn) {
      const int r = wn * 64 + fn * 16 + ln;
      bb[fn].u[0] = *(const uint4*)&Ws[r][lh * 8];
      bb[fn].u[1] = *(const uint4*)&Ws[r][16 + lh * 8];
    }
    #pragma unroll
    for (int fm = 0; fm < 2; ++fm)
      #pragma unroll
      for (int fn = 0; fn < 4; ++fn)
        acc[fm][fn] = wmma_bf16(a[fm], bb[fn], acc[fm][fn]);
  }

  #pragma unroll
  for (int fn = 0; fn < 4; ++fn) {
    const int nn = bn + wn * 64 + fn * 16 + ln;
    const float bv = bias[nn];
    const int hh = nn >> 6, dd = nn & 63;
    #pragma unroll
    for (int fm = 0; fm < 2; ++fm) {
      #pragma unroll
      for (int i = 0; i < 8; ++i) {
        const int mg = bm + wm * 32 + fm * 16 + i + 8 * lh;
        const int b = mg >> 11, s = mg & 2047;
        const float val = acc[fm][fn][i] + bv;
        const size_t idx = vtrans
            ? ((((size_t)b * NH_ + hh) * DH_ + dd) * S_ + s)
            : ((((size_t)b * NH_ + hh) * S_ + s) * DH_ + dd);
        out[idx] = f2bf(val);
      }
    }
  }
}

// ---------------------------------------------------------------------------
// Output projection: out[m,n] = Obf[m,:] . Wo[n,:] + bo[n], fp32 result.
// A operand is already bf16 -> raw copy: async global->LDS DMA (ASYNCcnt) with
// double-buffered LDS (one barrier per k-step); sync fallback if builtin absent.
// ---------------------------------------------------------------------------
__global__ __launch_bounds__(256) void out_proj_kernel(
    const unsigned short* __restrict__ Xb, const float* __restrict__ W,
    const float* __restrict__ bias, float* __restrict__ out)
{
  __shared__ __align__(16) unsigned short Xs[2][128][40];
  __shared__ __align__(16) unsigned short Ws[2][128][40];
  const int t = threadIdx.x;
  const int lane = t & 31, wave = t >> 5;
  const int wm = wave >> 1, wn = wave & 1;
  const int ln = lane & 15, lh = lane >> 4;
  const int bm = blockIdx.y * 128, bn = blockIdx.x * 128;

  const int row = t >> 1, seg = (t & 1) * 16;
  const unsigned short* xb = Xb + (size_t)(bm + row) * H_ + seg;
  const float* wb = W + (size_t)(bn + row) * H_ + seg;

  const v8f zero = {0.f,0.f,0.f,0.f,0.f,0.f,0.f,0.f};
  v8f acc[2][4];
  #pragma unroll
  for (int fm = 0; fm < 2; ++fm)
    #pragma unroll
    for (int fn = 0; fn < 4; ++fn) acc[fm][fn] = zero;

  float4 w0, w1, w2, w3;
#if !USE_ASYNC_LDS
  uint4 xv0, xv1;
#endif
  {                                    // prologue: stage tile 0 into buffer 0
#if USE_ASYNC_LDS
    async_ld16(xb,     &Xs[0][row][seg]);
    async_ld16(xb + 8, &Xs[0][row][seg + 8]);
#else
    const uint4* xp = (const uint4*)xb;
    *(uint4*)&Xs[0][row][seg]     = xp[0];
    *(uint4*)&Xs[0][row][seg + 8] = xp[1];
#endif
    const float4* wp = (const float4*)wb;
    w0 = wp[0]; w1 = wp[1]; w2 = wp[2]; w3 = wp[3];
    unsigned* dw = (unsigned*)&Ws[0][row][seg];
    dw[0] = pack2bf(w0.x, w0.y); dw[1] = pack2bf(w0.z, w0.w);
    dw[2] = pack2bf(w1.x, w1.y); dw[3] = pack2bf(w1.z, w1.w);
    dw[4] = pack2bf(w2.x, w2.y); dw[5] = pack2bf(w2.z, w2.w);
    dw[6] = pack2bf(w3.x, w3.y); dw[7] = pack2bf(w3.z, w3.w);
#if USE_ASYNC_LDS
    __builtin_amdgcn_s_wait_asynccnt(0);
#endif
  }

  for (int kt = 0; kt < H_; kt += 32) {
    const int cur = (kt >> 5) & 1, nxt = cur ^ 1;
    const bool more = (kt + 32) < H_;
    __syncthreads();                   // buffers[cur] ready; prev compute done
    if (more) {
#if USE_ASYNC_LDS
      async_ld16(xb + kt + 32,     &Xs[nxt][row][seg]);      // DMA overlaps WMMA
      async_ld16(xb + kt + 32 + 8, &Xs[nxt][row][seg + 8]);
#else
      const uint4* xp = (const uint4*)(xb + kt + 32);
      xv0 = xp[0]; xv1 = xp[1];
#endif
      const float4* wp = (const float4*)(wb + kt + 32);
      w0 = wp[0]; w1 = wp[1]; w2 = wp[2]; w3 = wp[3];
    }
    BF16x16 a[2], bb[4];
    #pragma unroll
    for (int fm = 0; fm < 2; ++fm) {
      const int r = wm * 32 + fm * 16 + ln;
      a[fm].u[0] = *(const uint4*)&Xs[cur][r][lh * 8];
      a[fm].u[1] = *(const uint4*)&Xs[cur][r][16 + lh * 8];
    }
    #pragma unroll
    for (int fn = 0; fn < 4; ++fn) {
      const int r = wn * 64 + fn * 16 + ln;
      bb[fn].u[0] = *(const uint4*)&Ws[cur][r][lh * 8];
      bb[fn].u[1] = *(const uint4*)&Ws[cur][r][16 + lh * 8];
    }
    #pragma unroll
    for (int fm = 0; fm < 2; ++fm)
      #pragma unroll
      for (int fn = 0; fn < 4; ++fn)
        acc[fm][fn] = wmma_bf16(a[fm], bb[fn], acc[fm][fn]);
    if (more) {
      unsigned* dw = (unsigned*)&Ws[nxt][row][seg];
      dw[0] = pack2bf(w0.x, w0.y); dw[1] = pack2bf(w0.z, w0.w);
      dw[2] = pack2bf(w1.x, w1.y); dw[3] = pack2bf(w1.z, w1.w);
      dw[4] = pack2bf(w2.x, w2.y); dw[5] = pack2bf(w2.z, w2.w);
      dw[6] = pack2bf(w3.x, w3.y); dw[7] = pack2bf(w3.z, w3.w);
#if !USE_ASYNC_LDS
      *(uint4*)&Xs[nxt][row][seg]     = xv0;
      *(uint4*)&Xs[nxt][row][seg + 8] = xv1;
#endif
#if USE_ASYNC_LDS
      __builtin_amdgcn_s_wait_asynccnt(0);
#endif
    }
  }

  #pragma unroll
  for (int fn = 0; fn < 4; ++fn) {
    const int nn = bn + wn * 64 + fn * 16 + ln;
    const float bv = bias[nn];
    #pragma unroll
    for (int fm = 0; fm < 2; ++fm) {
      #pragma unroll
      for (int i = 0; i < 8; ++i) {
        const int mg = bm + wm * 32 + fm * 16 + i + 8 * lh;
        out[(size_t)mg * H_ + nn] = acc[fm][fn][i] + bv;
      }
    }
  }
}

// ---------------------------------------------------------------------------
// Flash attention, causal + padding mask, FILL semantics matching reference.
// Grid: (B*NH, S/64). Block: 128 threads = 4 waves, each wave owns 16 rows.
// Q,K: [B,NH,S,DH] bf16. V: [B,NH,DH,S] bf16 (pre-transposed). O: [B,S,H] bf16.
// ---------------------------------------------------------------------------
__global__ __launch_bounds__(128) void attn_kernel(
    const unsigned short* __restrict__ Qb, const unsigned short* __restrict__ Kb,
    const unsigned short* __restrict__ Vt, const float* __restrict__ mask,
    unsigned short* __restrict__ Ob)
{
  __shared__ __align__(16) unsigned short Psh[4][16][32];
  const int lane = threadIdx.x & 31, wave = threadIdx.x >> 5;
  const int ln = lane & 15, lh = lane >> 4;
  const int bh = blockIdx.x;
  const int b = bh >> 4, h = bh & 15;
  const int q0 = blockIdx.y * 64;
  const size_t hoff = (size_t)bh * S_ * DH_;
  const unsigned short* Qh = Qb + hoff;
  const unsigned short* Kh = Kb + hoff;
  const unsigned short* Vh = Vt + hoff;   // [DH_][S_]

  const int qrow = q0 + wave * 16 + ln;
  BF16x16 aq[2];
  #pragma unroll
  for (int st = 0; st < 2; ++st) {
    aq[st].u[0] = *(const uint4*)(Qh + (size_t)qrow * DH_ + st * 32 + lh * 8);
    aq[st].u[1] = *(const uint4*)(Qh + (size_t)qrow * DH_ + st * 32 + 16 + lh * 8);
  }

  const v8f zero = {0.f,0.f,0.f,0.f,0.f,0.f,0.f,0.f};
  v8f oacc[4] = {zero, zero, zero, zero};
  float mrun[8], lrun[8];
  #pragma unroll
  for (int i = 0; i < 8; ++i) { mrun[i] = -3.0e38f; lrun[i] = 0.f; }

  const int qbase = q0 + wave * 16 + 8 * lh;
  const int jtiles = (q0 >> 5) + 2;

  for (int jt = 0; jt < jtiles; ++jt) {
    const int j0 = jt * 32;
    v8f sc[2];
    #pragma unroll
    for (int fn = 0; fn < 2; ++fn) {
      v8f cz = zero;
      #pragma unroll
      for (int st = 0; st < 2; ++st) {
        BF16x16 bk;
        const unsigned short* kp = Kh + (size_t)(j0 + fn * 16 + ln) * DH_ + st * 32;
        bk.u[0] = *(const uint4*)(kp + lh * 8);
        bk.u[1] = *(const uint4*)(kp + 16 + lh * 8);
        cz = wmma_bf16(aq[st], bk, cz);
      }
      sc[fn] = cz;
    }
    #pragma unroll
    for (int fn = 0; fn < 2; ++fn) {
      const int key = j0 + fn * 16 + ln;
      const float mk = mask[(size_t)b * S_ + key];
      #pragma unroll
      for (int i = 0; i < 8; ++i) {
        const float sv = sc[fn][i] * 0.125f;   // 1/sqrt(64)
        sc[fn][i] = ((key <= qbase + i) && (mk != 0.f)) ? sv : FILLV;
      }
    }
    #pragma unroll
    for (int i = 0; i < 8; ++i) {
      float mn = rmax16(fmaxf(sc[0][i], sc[1][i]));
      const float mt = fmaxf(mrun[i], mn);
      const float alpha = __expf(mrun[i] - mt);
      mrun[i] = mt;
      const float p0 = __expf(sc[0][i] - mt);
      const float p1 = __expf(sc[1][i] - mt);
      sc[0][i] = p0; sc[1][i] = p1;
      lrun[i] = lrun[i] * alpha + rsum16(p0 + p1);
      #pragma unroll
      for (int nf = 0; nf < 4; ++nf) oacc[nf][i] *= alpha;
    }
    #pragma unroll
    for (int i = 0; i < 8; ++i) {
      Psh[wave][i + 8 * lh][ln]      = f2bf(sc[0][i]);
      Psh[wave][i + 8 * lh][16 + ln] = f2bf(sc[1][i]);
    }
#if __has_builtin(__builtin_amdgcn_wave_barrier)
    __builtin_amdgcn_wave_barrier();
#else
    __syncthreads();
#endif
    BF16x16 pa;
    pa.u[0] = *(const uint4*)&Psh[wave][ln][lh * 8];
    pa.u[1] = *(const uint4*)&Psh[wave][ln][16 + lh * 8];
#if __has_builtin(__builtin_amdgcn_wave_barrier)
    __builtin_amdgcn_wave_barrier();
#else
    __syncthreads();
#endif
    #pragma unroll
    for (int nf = 0; nf < 4; ++nf) {
      BF16x16 bv;
      const unsigned short* vp = Vh + (size_t)(nf * 16 + ln) * S_ + j0;
      bv.u[0] = *(const uint4*)(vp + lh * 8);
      bv.u[1] = *(const uint4*)(vp + 16 + lh * 8);
      oacc[nf] = wmma_bf16(pa, bv, oacc[nf]);
    }
  }

  #pragma unroll
  for (int i = 0; i < 8; ++i) {
    const float inv = 1.0f / lrun[i];
    const int qi = qbase + i;
    #pragma unroll
    for (int nf = 0; nf < 4; ++nf) {
      Ob[((size_t)b * S_ + qi) * H_ + h * DH_ + nf * 16 + ln] = f2bf(oacc[nf][i] * inv);
    }
  }
}

// ---------------------------------------------------------------------------
extern "C" void kernel_launch(void* const* d_in, const int* in_sizes, int n_in,
                              void* d_out, int out_size, void* d_ws, size_t ws_size,
                              hipStream_t stream) {
  const float* q   = (const float*)d_in[0];
  const float* k   = (const float*)d_in[1];
  const float* v   = (const float*)d_in[2];
  const float* msk = (const float*)d_in[3];
  const float* Wq  = (const float*)d_in[4];
  const float* bq  = (const float*)d_in[5];
  const float* Wk  = (const float*)d_in[6];
  const float* bk  = (const float*)d_in[7];
  const float* Wv  = (const float*)d_in[8];
  const float* bv  = (const float*)d_in[9];
  const float* Wo  = (const float*)d_in[10];
  const float* bo  = (const float*)d_in[11];
  (void)in_sizes; (void)n_in; (void)out_size; (void)ws_size;

  const size_t perT = (size_t)B_ * NH_ * S_ * DH_;   // 8M elems (bf16)
  unsigned short* Qw = (unsigned short*)d_ws;
  unsigned short* Kw = Qw + perT;
  unsigned short* Vw = Kw + perT;
  unsigned short* Ow = Vw + perT;

  dim3 gg(H_ / 128, (B_ * S_) / 128);                // 8 x 64 blocks
  qkv_proj_kernel<<<gg, 256, 0, stream>>>(q, Wq, bq, Qw, 0);
  qkv_proj_kernel<<<gg, 256, 0, stream>>>(k, Wk, bk, Kw, 0);
  qkv_proj_kernel<<<gg, 256, 0, stream>>>(v, Wv, bv, Vw, 1);
  attn_kernel<<<dim3(B_ * NH_, S_ / 64), 128, 0, stream>>>(Qw, Kw, Vw, msk, Ow);
  out_proj_kernel<<<gg, 256, 0, stream>>>(Ow, Wo, bo, (float*)d_out);
}
